// TreeSummarize_71528385348002
// MI455X (gfx1250) — matrix operand order repaired and checked by reading, not compiled, not run on hardware
//
#include <hip/hip_runtime.h>
#include <hip/hip_bf16.h>
#include <math.h>

typedef __attribute__((ext_vector_type(16))) __bf16 v16bf;
typedef __attribute__((ext_vector_type(8)))  float  v8f;

constexpr int N_  = 2048;
constexpr int D_  = 1024;
constexpr int M_  = 1024;
constexpr int M3  = 3 * M_;
constexpr int M4  = 4 * M_;
constexpr int MAX_LVL = 96;   // random increasing tree height ~e*ln(N) ~ 21; 96 is very safe

union ABu { int4 q[2]; v16bf v; };

__device__ __forceinline__ float sigmoidf_(float x) { return 1.f / (1.f + __expf(-x)); }

#define WMMA_BF16(A, B, C) \
  __builtin_amdgcn_wmma_f32_16x16x32_bf16(false, (A), false, (B), (short)0, (C), false, false)

// ---------------- init / conversion ----------------

__global__ void k_zero(float* __restrict__ h_sum, float* __restrict__ fc_sum,
                       float* __restrict__ fx_all) {
  size_t total = (size_t)(N_ + 1) * M_;
  size_t stride = (size_t)gridDim.x * blockDim.x;
  for (size_t i = (size_t)blockIdx.x * blockDim.x + threadIdx.x; i < total; i += stride) {
    h_sum[i]  = 0.f;
    fc_sum[i] = 0.f;
    if (i >= (size_t)N_ * M_) fx_all[i] = 0.f;   // dummy row N (fx of root's "parent")
  }
}

__global__ void k_cvt_bf16(const float* __restrict__ s, __bf16* __restrict__ d, int n) {
  int stride = gridDim.x * blockDim.x;
  for (int i = blockIdx.x * blockDim.x + threadIdx.x; i < n; i += stride)
    d[i] = (__bf16)s[i];
}

// ---------------- level schedule (O(N), single thread, deterministic) ----------------

__global__ void k_levels(const int* __restrict__ parent, int* __restrict__ lvl,
                         int* __restrict__ order, int* __restrict__ meta) {
  if (threadIdx.x != 0 || blockIdx.x != 0) return;
  for (int t = 0; t < N_; t++) lvl[t] = 0;
  for (int t = 0; t < N_ - 1; t++) {           // parent[t] > t, so one forward pass suffices
    int p = parent[t];
    if (p >= 0 && p < N_) {
      int c = lvl[t] + 1;
      if (c > MAX_LVL - 1) c = MAX_LVL - 1;
      if (c > lvl[p]) lvl[p] = c;
    }
  }
  int cnt[MAX_LVL];
  for (int l = 0; l < MAX_LVL; l++) cnt[l] = 0;
  for (int t = 0; t < N_; t++) cnt[lvl[t]]++;
  int s = 0;
  for (int l = 0; l < MAX_LVL; l++) { meta[l] = s; s += cnt[l]; }
  meta[MAX_LVL] = s;
  int pos[MAX_LVL];
  for (int l = 0; l < MAX_LVL; l++) pos[l] = meta[l];
  for (int t = 0; t < N_; t++) order[pos[lvl[t]]++] = t;
}

// ---------------- upfront GEMM: [ioux_all | fx_all] = X(bf16) * W1(bf16)^T + bias ----------------
// One wave computes a 32x32 f32 tile (2x2 WMMA register blocking -> 4 wmma per 8 b128 loads).

__global__ void __launch_bounds__(256) k_gemm_x(
    const __bf16* __restrict__ X, const __bf16* __restrict__ W1,
    const float* __restrict__ bioux, const float* __restrict__ bfx,
    float* __restrict__ ioux, float* __restrict__ fx)
{
  const int lane = threadIdx.x & 31;
  const int wave = (blockIdx.x * blockDim.x + threadIdx.x) >> 5;
  const int nw   = (gridDim.x * blockDim.x) >> 5;
  const int r  = lane & 15;
  const int hi = lane >> 4;
  const int Mt2 = N_ / 32, Nt2 = M4 / 32;
  for (int t = wave; t < Mt2 * Nt2; t += nw) {
    int mt = t / Nt2, nt = t % Nt2;
    const __bf16* Ap0 = X  + (size_t)(mt * 32 + r) * D_;
    const __bf16* Ap1 = Ap0 + (size_t)16 * D_;
    const __bf16* Wp0 = W1 + (size_t)(nt * 32 + r) * D_;
    const __bf16* Wp1 = Wp0 + (size_t)16 * D_;
    v8f c00 = {}, c01 = {}, c10 = {}, c11 = {};
    for (int k0 = 0; k0 < D_; k0 += 32) {
      ABu a0, a1, b0, b1;
      // A 16x32 bf16: lane(L&15)=row; K chunks [hi*8..+7] and [16+hi*8..+7]
      a0.q[0] = *(const int4*)(Ap0 + k0 + hi * 8);
      a0.q[1] = *(const int4*)(Ap0 + k0 + 16 + hi * 8);
      a1.q[0] = *(const int4*)(Ap1 + k0 + hi * 8);
      a1.q[1] = *(const int4*)(Ap1 + k0 + 16 + hi * 8);
      // B 32x16 bf16: lane holds col (L&15), K chunks [hi*8..+7] and [16+hi*8..+7]
      b0.q[0] = *(const int4*)(Wp0 + k0 + hi * 8);
      b0.q[1] = *(const int4*)(Wp0 + k0 + 16 + hi * 8);
      b1.q[0] = *(const int4*)(Wp1 + k0 + hi * 8);
      b1.q[1] = *(const int4*)(Wp1 + k0 + 16 + hi * 8);
      c00 = WMMA_BF16(a0.v, b0.v, c00);
      c01 = WMMA_BF16(a0.v, b1.v, c01);
      c10 = WMMA_BF16(a1.v, b0.v, c10);
      c11 = WMMA_BF16(a1.v, b1.v, c11);
    }
    const int mb = mt * 32, nb = nt * 32;
#pragma unroll
    for (int q = 0; q < 4; q++) {
      v8f& acc = (q == 0) ? c00 : (q == 1) ? c01 : (q == 2) ? c10 : c11;
      int rowb = mb + ((q >> 1) << 4);
      int col  = nb + ((q & 1) << 4) + r;
#pragma unroll
      for (int v = 0; v < 8; v++) {
        int row = rowb + v + 8 * hi;    // C layout: VGPR v -> M = v + 8*(lane>>4)
        if (col < M3) ioux[(size_t)row * M3 + col]        = acc[v] + bioux[col];
        else          fx  [(size_t)row * M_ + (col - M3)] = acc[v] + bfx[col - M3];
      }
    }
  }
}

// ---------------- per-level GEMM 1: ioux_all[node] += h_sum[node] * Wiouh^T ----------------
// A rows gathered through order[] from fp32 h_sum, converted to bf16 in-register.

__global__ void __launch_bounds__(128) k_gemm_iouh(
    const float* __restrict__ h_sum, const __bf16* __restrict__ Wiouh,
    const int* __restrict__ order, const int* __restrict__ meta, int l,
    float* __restrict__ ioux)
{
  int ls = meta[l]; int B = meta[l + 1] - ls;
  if (B <= 0) return;
  const int lane = threadIdx.x & 31;
  const int wave = (blockIdx.x * blockDim.x + threadIdx.x) >> 5;
  const int nw   = (gridDim.x * blockDim.x) >> 5;
  const int r  = lane & 15;
  const int hi = lane >> 4;
  const int Bt2 = (B + 31) >> 5, Nt2 = M3 / 32;
  for (int t = wave; t < Bt2 * Nt2; t += nw) {
    int mt = t / Nt2, nt = t % Nt2;
    int m0 = mt * 32 + r;      if (m0 >= B) m0 = B - 1;   // clamp; guarded at write
    int m1 = mt * 32 + 16 + r; if (m1 >= B) m1 = B - 1;
    const float*  Hp0 = h_sum + (size_t)order[ls + m0] * M_;
    const float*  Hp1 = h_sum + (size_t)order[ls + m1] * M_;
    const __bf16* Wp0 = Wiouh + (size_t)(nt * 32 + r) * M_;
    const __bf16* Wp1 = Wp0 + (size_t)16 * M_;
    v8f c00 = {}, c01 = {}, c10 = {}, c11 = {};
    for (int k0 = 0; k0 < M_; k0 += 32) {
      v16bf a0, a1;
#pragma unroll
      for (int i = 0; i < 8; i++) {
        a0[i]     = (__bf16)Hp0[k0 + hi * 8 + i];
        a0[8 + i] = (__bf16)Hp0[k0 + 16 + hi * 8 + i];
        a1[i]     = (__bf16)Hp1[k0 + hi * 8 + i];
        a1[8 + i] = (__bf16)Hp1[k0 + 16 + hi * 8 + i];
      }
      ABu b0, b1;
      b0.q[0] = *(const int4*)(Wp0 + k0 + hi * 8);
      b0.q[1] = *(const int4*)(Wp0 + k0 + 16 + hi * 8);
      b1.q[0] = *(const int4*)(Wp1 + k0 + hi * 8);
      b1.q[1] = *(const int4*)(Wp1 + k0 + 16 + hi * 8);
      c00 = WMMA_BF16(a0, b0.v, c00);
      c01 = WMMA_BF16(a0, b1.v, c01);
      c10 = WMMA_BF16(a1, b0.v, c10);
      c11 = WMMA_BF16(a1, b1.v, c11);
    }
    const int mb = mt * 32, nb = nt * 32;
#pragma unroll
    for (int q = 0; q < 4; q++) {
      v8f& acc = (q == 0) ? c00 : (q == 1) ? c01 : (q == 2) ? c10 : c11;
      int rowb = mb + ((q >> 1) << 4);
      int col  = nb + ((q & 1) << 4) + r;
#pragma unroll
      for (int v = 0; v < 8; v++) {
        int row = rowb + v + 8 * hi;
        if (row < B) ioux[(size_t)order[ls + row] * M3 + col] += acc[v];
      }
    }
  }
}

// ---------------- per-level gates: i,o,u -> c,h ; scatter h into parent ----------------

__global__ void __launch_bounds__(256) k_gates(
    const int* __restrict__ order, const int* __restrict__ parent,
    const int* __restrict__ meta, int l,
    const float* __restrict__ ioux, const float* __restrict__ biouh,
    const float* __restrict__ fc_sum, float* __restrict__ h_sum,
    float* __restrict__ c_lev, __bf16* __restrict__ Hh, float* __restrict__ out)
{
  int ls = meta[l]; int B = meta[l + 1] - ls;
  if (B <= 0) return;
  int total  = B * M_;
  int stride = gridDim.x * blockDim.x;
  for (int e = blockIdx.x * blockDim.x + threadIdx.x; e < total; e += stride) {
    int m = e >> 10;           // / M_
    int j = e & (M_ - 1);
    int node = order[ls + m];
    const float* row = ioux + (size_t)node * M3;
    float iv = sigmoidf_(row[j]          + biouh[j]);
    float ov = sigmoidf_(row[M_ + j]     + biouh[M_ + j]);
    float uv = tanhf   (row[2 * M_ + j]  + biouh[2 * M_ + j]);
    float c  = iv * uv + fc_sum[(size_t)node * M_ + j];
    float h  = ov * tanhf(c);
    c_lev[e] = c;                         // compact [m, j] for this level
    Hh[e]    = (__bf16)h;                 // compact bf16 h, A-matrix for Wfh GEMM
    atomicAdd(&h_sum[(size_t)parent[node] * M_ + j], h);
    if (node == N_ - 1) { out[j] = c; out[M_ + j] = h; }   // root outputs (c, h)
  }
}

// ---------------- per-level GEMM 2: f = sigmoid(h*Wfh^T + bfh + fx[parent]); fc_sum[parent] += f*c --

__global__ void __launch_bounds__(128) k_gemm_fh(
    const __bf16* __restrict__ Hh, const __bf16* __restrict__ Wfh,
    const float* __restrict__ bfh, const float* __restrict__ fx_all,
    const float* __restrict__ c_lev, const int* __restrict__ order,
    const int* __restrict__ parent, const int* __restrict__ meta, int l,
    float* __restrict__ fc_sum)
{
  int ls = meta[l]; int B = meta[l + 1] - ls;
  if (B <= 0) return;
  const int lane = threadIdx.x & 31;
  const int wave = (blockIdx.x * blockDim.x + threadIdx.x) >> 5;
  const int nw   = (gridDim.x * blockDim.x) >> 5;
  const int r  = lane & 15;
  const int hi = lane >> 4;
  const int Bt2 = (B + 31) >> 5, Nt2 = M_ / 32;
  for (int t = wave; t < Bt2 * Nt2; t += nw) {
    int mt = t / Nt2, nt = t % Nt2;
    int m0 = mt * 32 + r;      if (m0 >= B) m0 = B - 1;
    int m1 = mt * 32 + 16 + r; if (m1 >= B) m1 = B - 1;
    const __bf16* Ap0 = Hh  + (size_t)m0 * M_;            // compact bf16 h rows
    const __bf16* Ap1 = Hh  + (size_t)m1 * M_;
    const __bf16* Wp0 = Wfh + (size_t)(nt * 32 + r) * M_;
    const __bf16* Wp1 = Wp0 + (size_t)16 * M_;
    v8f c00 = {}, c01 = {}, c10 = {}, c11 = {};
    for (int k0 = 0; k0 < M_; k0 += 32) {
      ABu a0, a1, b0, b1;
      a0.q[0] = *(const int4*)(Ap0 + k0 + hi * 8);
      a0.q[1] = *(const int4*)(Ap0 + k0 + 16 + hi * 8);
      a1.q[0] = *(const int4*)(Ap1 + k0 + hi * 8);
      a1.q[1] = *(const int4*)(Ap1 + k0 + 16 + hi * 8);
      b0.q[0] = *(const int4*)(Wp0 + k0 + hi * 8);
      b0.q[1] = *(const int4*)(Wp0 + k0 + 16 + hi * 8);
      b1.q[0] = *(const int4*)(Wp1 + k0 + hi * 8);
      b1.q[1] = *(const int4*)(Wp1 + k0 + 16 + hi * 8);
      c00 = WMMA_BF16(a0.v, b0.v, c00);
      c01 = WMMA_BF16(a0.v, b1.v, c01);
      c10 = WMMA_BF16(a1.v, b0.v, c10);
      c11 = WMMA_BF16(a1.v, b1.v, c11);
    }
    const int mb = mt * 32, nb = nt * 32;
#pragma unroll
    for (int q = 0; q < 4; q++) {
      v8f& acc = (q == 0) ? c00 : (q == 1) ? c01 : (q == 2) ? c10 : c11;
      int rowb = mb + ((q >> 1) << 4);
      int col  = nb + ((q & 1) << 4) + r;
#pragma unroll
      for (int v = 0; v < 8; v++) {
        int row = rowb + v + 8 * hi;
        if (row < B) {
          int node = order[ls + row];
          int p    = parent[node];                        // p may be N_ (dummy row, zeroed)
          float pre = acc[v] + bfh[col] + fx_all[(size_t)p * M_ + col];
          float f   = sigmoidf_(pre);
          atomicAdd(&fc_sum[(size_t)p * M_ + col], f * c_lev[(size_t)row * M_ + col]);
        }
      }
    }
  }
}

// ---------------- host ----------------

extern "C" void kernel_launch(void* const* d_in, const int* in_sizes, int n_in,
                              void* d_out, int out_size, void* d_ws, size_t ws_size,
                              hipStream_t stream) {
  (void)in_sizes; (void)n_in; (void)out_size; (void)ws_size;
  const float* inputs = (const float*)d_in[0];
  const int*   parent = (const int*)  d_in[1];
  const float* Wioux  = (const float*)d_in[2];
  const float* bioux  = (const float*)d_in[3];
  const float* Wiouh  = (const float*)d_in[4];
  const float* biouh  = (const float*)d_in[5];
  const float* Wfx    = (const float*)d_in[6];
  const float* bfx    = (const float*)d_in[7];
  const float* Wfh    = (const float*)d_in[8];
  const float* bfh    = (const float*)d_in[9];
  float* out = (float*)d_out;

  char* w = (char*)d_ws;
  auto alloc = [&](size_t bytes) {
    char* p = w;
    w += (bytes + 255) & ~(size_t)255;
    return p;
  };
  float*  ioux_all = (float*) alloc((size_t)N_ * M3 * 4);        // 24 MB
  float*  fx_all   = (float*) alloc((size_t)(N_ + 1) * M_ * 4);  // 8.4 MB (row N zero)
  float*  h_sum    = (float*) alloc((size_t)(N_ + 1) * M_ * 4);  // 8.4 MB
  float*  fc_sum   = (float*) alloc((size_t)(N_ + 1) * M_ * 4);  // 8.4 MB
  float*  c_lev    = (float*) alloc((size_t)N_ * M_ * 4);        // 8 MB
  __bf16* Hh       = (__bf16*)alloc((size_t)N_ * M_ * 2);        // 4 MB
  __bf16* Xb       = (__bf16*)alloc((size_t)N_ * D_ * 2);        // 4 MB
  __bf16* W1b      = (__bf16*)alloc((size_t)M4 * D_ * 2);        // 8 MB  [Wioux;Wfx]
  __bf16* Wiouhb   = (__bf16*)alloc((size_t)M3 * M_ * 2);        // 6 MB
  __bf16* Wfhb     = (__bf16*)alloc((size_t)M_ * M_ * 2);        // 2 MB
  int*    lvl      = (int*)   alloc((size_t)N_ * 4);
  int*    order    = (int*)   alloc((size_t)N_ * 4);
  int*    meta     = (int*)   alloc((size_t)(MAX_LVL + 1) * 4);

  k_zero<<<512, 256, 0, stream>>>(h_sum, fc_sum, fx_all);
  k_cvt_bf16<<<512, 256, 0, stream>>>(inputs, Xb, N_ * D_);
  k_cvt_bf16<<<512, 256, 0, stream>>>(Wioux, W1b, M3 * D_);
  k_cvt_bf16<<<512, 256, 0, stream>>>(Wfx, W1b + (size_t)M3 * D_, M_ * D_);
  k_cvt_bf16<<<512, 256, 0, stream>>>(Wiouh, Wiouhb, M3 * M_);
  k_cvt_bf16<<<512, 256, 0, stream>>>(Wfh, Wfhb, M_ * M_);
  k_levels<<<1, 1, 0, stream>>>(parent, lvl, order, meta);
  k_gemm_x<<<1024, 256, 0, stream>>>(Xb, W1b, bioux, bfx, ioux_all, fx_all);

  for (int l = 0; l < MAX_LVL; l++) {
    if (l > 0)   // level 0 = leaves: h_sum rows are zero, Wiouh GEMM contributes nothing
      k_gemm_iouh<<<512, 128, 0, stream>>>(h_sum, Wiouhb, order, meta, l, ioux_all);
    k_gates<<<256, 256, 0, stream>>>(order, parent, meta, l, ioux_all, biouh,
                                     fc_sum, h_sum, c_lev, Hh, out);
    k_gemm_fh<<<512, 128, 0, stream>>>(Hh, Wfhb, bfh, fx_all, c_lev, order,
                                       parent, meta, l, fc_sum);
  }
}
